// Block_37804302139511
// MI455X (gfx1250) — matrix operand order repaired and checked by reading, not compiled, not run on hardware
//
#include <hip/hip_runtime.h>

// ---------------------------------------------------------------------------
// Transformer decoder block (pre-LN) for MI455X / gfx1250, wave32 + WMMA.
// All GEMMs and attention matmuls use v_wmma_f32_16x16x32_bf16.
// Attention K-tiles staged via global_load_async_to_lds_b128 (ASYNCcnt).
// ---------------------------------------------------------------------------

typedef __bf16 bf16_t;
typedef __attribute__((ext_vector_type(16))) __bf16 v16bf;
typedef __attribute__((ext_vector_type(8)))  float  v8f;

#define B_  4
#define T_  2048
#define E_  512
#define H_  8
#define D_  64
#define HID_ 4096
#define M_  (B_ * T_)   // 8192 rows

union FragU { v16bf v; uint4 q[2]; };

// A-fragment (16x32 bf16, M x K): lane -> M=(l&15); elements are K = e + 16*hi,
// i.e. 16 CONTIGUOUS K values. p = rowptr + hi*16 (+k).
__device__ __forceinline__ v16bf load_a_frag(const bf16_t* p) {
  FragU f;
  f.q[0] = *(const uint4*)(p);
  f.q[1] = *(const uint4*)(p + 8);
  return f.v;
}

// B-fragment (32x16 bf16, K x N): lane -> N=(l&15); elements are K = e + 16*hi,
// i.e. 16 CONTIGUOUS K values. Source stored as Bt[N][K]; p = Nrowptr + hi*16 (+k).
__device__ __forceinline__ v16bf load_b_frag(const bf16_t* p) {
  FragU f;
  f.q[0] = *(const uint4*)(p);
  f.q[1] = *(const uint4*)(p + 8);
  return f.v;
}

__device__ __forceinline__ v8f wmma_bf16(v16bf a, v16bf b, v8f c) {
  return __builtin_amdgcn_wmma_f32_16x16x32_bf16(false, a, false, b, (short)0, c,
                                                 false, false);
}

// ---------------------------------------------------------------------------
// Weight conversion: Wq/Wk/Wv [H,E,D] f32 -> Wqkvt [(part*512 + h*64 + d)][e] bf16
// ---------------------------------------------------------------------------
__global__ __launch_bounds__(256) void cvt_qkv_w(const float* __restrict__ Wq,
                                                 const float* __restrict__ Wk,
                                                 const float* __restrict__ Wv,
                                                 bf16_t* __restrict__ dst) {
  int idx = blockIdx.x * 256 + threadIdx.x;   // 3*512*512 total
  int e = idx & 511;
  int r = idx >> 9;             // 0..1535
  int d = r & 63;
  int h = (r >> 6) & 7;
  int part = r >> 9;
  const float* W = (part == 0) ? Wq : (part == 1) ? Wk : Wv;
  dst[(size_t)r * 512 + e] = (bf16_t)W[((size_t)h * 512 + e) * 64 + d];
}

// Generic: dst[n*K + k] = (bf16) src[k*N + n]     (src is [K rows][N cols] f32)
__global__ __launch_bounds__(256) void transpose_cvt(const float* __restrict__ src,
                                                     bf16_t* __restrict__ dst,
                                                     int N, int K) {
  size_t idx = (size_t)blockIdx.x * 256 + threadIdx.x;
  if (idx >= (size_t)N * K) return;
  int k = (int)(idx % K);
  int n = (int)(idx / K);
  dst[(size_t)n * K + k] = (bf16_t)src[(size_t)k * N + n];
}

// ---------------------------------------------------------------------------
// LayerNorm: one wave per row of 512. Emits f32 and bf16 copies.
// ---------------------------------------------------------------------------
__global__ __launch_bounds__(256) void ln_kernel(const float* __restrict__ x,
                                                 const float* __restrict__ g,
                                                 const float* __restrict__ bta,
                                                 float* __restrict__ y,
                                                 bf16_t* __restrict__ yb) {
  const int lane = threadIdx.x & 31;
  const int wave = threadIdx.x >> 5;
  const size_t row = (size_t)blockIdx.x * 8 + wave;
  const float* xr = x + row * E_;
  float4 v[4];
  float sum = 0.f, sq = 0.f;
#pragma unroll
  for (int i = 0; i < 4; ++i) {
    v[i] = *(const float4*)(xr + i * 128 + lane * 4);
    sum += v[i].x + v[i].y + v[i].z + v[i].w;
    sq  += v[i].x * v[i].x + v[i].y * v[i].y + v[i].z * v[i].z + v[i].w * v[i].w;
  }
#pragma unroll
  for (int off = 1; off < 32; off <<= 1) {
    sum += __shfl_xor(sum, off, 32);
    sq  += __shfl_xor(sq, off, 32);
  }
  const float mean = sum * (1.0f / E_);
  const float var  = sq * (1.0f / E_) - mean * mean;
  const float rstd = rsqrtf(var + 1e-5f);
  float*  yr  = y  + row * E_;
  bf16_t* ybr = yb + row * E_;
#pragma unroll
  for (int i = 0; i < 4; ++i) {
    const int c = i * 128 + lane * 4;
    float4 gg = *(const float4*)(g + c);
    float4 bb = *(const float4*)(bta + c);
    float4 o;
    o.x = (v[i].x - mean) * rstd * gg.x + bb.x;
    o.y = (v[i].y - mean) * rstd * gg.y + bb.y;
    o.z = (v[i].z - mean) * rstd * gg.z + bb.z;
    o.w = (v[i].w - mean) * rstd * gg.w + bb.w;
    *(float4*)(yr + c) = o;
    ybr[c + 0] = (bf16_t)o.x;
    ybr[c + 1] = (bf16_t)o.y;
    ybr[c + 2] = (bf16_t)o.z;
    ybr[c + 3] = (bf16_t)o.w;
  }
}

// ---------------------------------------------------------------------------
// Generic WMMA GEMM: C[M,N] = A_bf[M,K] * Bt_bf[N,K]^T  (+bias)(+ReLU)(+res)
// Wave -> 32x64 output tile (8 WMMAs / 12 b128 loads per k-step);
// block (128 thr = 4 waves) -> 128x64.
// ---------------------------------------------------------------------------
template <int HAS_BIAS, int RELU, int HAS_RES, int OUT_BF>
__global__ __launch_bounds__(128)
void gemm_kernel(const bf16_t* __restrict__ A, const bf16_t* __restrict__ Bt,
                 const float* __restrict__ bias, const float* __restrict__ res,
                 float* __restrict__ outf, bf16_t* __restrict__ outb,
                 int K, int lda, int ldb, int ldo) {
  const int lane = threadIdx.x & 31;
  const int wave = threadIdx.x >> 5;
  const int hi   = (lane >> 4) & 1;
  const int l15  = lane & 15;
  const int mrow = blockIdx.y * 128 + wave * 32;
  const int ncol = blockIdx.x * 64;

  const bf16_t* ap0 = A + (size_t)(mrow + l15) * lda + hi * 16;
  const bf16_t* ap1 = ap0 + (size_t)16 * lda;
  const bf16_t* bp[4];
#pragma unroll
  for (int c = 0; c < 4; ++c)
    bp[c] = Bt + (size_t)(ncol + c * 16 + l15) * ldb + hi * 16;

  v8f acc[2][4];
#pragma unroll
  for (int r = 0; r < 2; ++r)
#pragma unroll
    for (int c = 0; c < 4; ++c) acc[r][c] = (v8f){0, 0, 0, 0, 0, 0, 0, 0};

  for (int k = 0; k < K; k += 32) {
    v16bf a0 = load_a_frag(ap0 + k);
    v16bf a1 = load_a_frag(ap1 + k);
#pragma unroll
    for (int c = 0; c < 4; ++c) {
      v16bf b = load_b_frag(bp[c] + k);
      acc[0][c] = wmma_bf16(a0, b, acc[0][c]);
      acc[1][c] = wmma_bf16(a1, b, acc[1][c]);
    }
  }

#pragma unroll
  for (int r = 0; r < 2; ++r) {
#pragma unroll
    for (int c = 0; c < 4; ++c) {
      const int col = ncol + c * 16 + l15;
      const float bs = HAS_BIAS ? bias[col] : 0.0f;
#pragma unroll
      for (int e = 0; e < 8; ++e) {
        const int row = mrow + r * 16 + e + hi * 8;
        float v = acc[r][c][e] + bs;
        if (RELU) v = fmaxf(v, 0.0f);
        const size_t idx = (size_t)row * ldo + col;
        if (HAS_RES) v += res[idx];
        if (OUT_BF) outb[idx] = (bf16_t)v;
        else        outf[idx] = v;
      }
    }
  }
}

// ---------------------------------------------------------------------------
// Flash-style causal attention, scale = T^-0.5.
// qkv layout: [B*T][1536] bf16 (q|k|v each H*D). Block: (b,h, 128 q-rows).
// K tile staged with async global->LDS DMA; V staged+transposed through VGPRs.
// ---------------------------------------------------------------------------
__global__ __launch_bounds__(256)
void attn_kernel(const bf16_t* __restrict__ qkv, bf16_t* __restrict__ obuf) {
  __shared__ __align__(16) bf16_t Ks[32 * 64];      // K block [key][d]
  __shared__ __align__(16) bf16_t Vt[64 * 32];      // V block transposed [d][key]
  __shared__ __align__(16) bf16_t Ps[8][16 * 32];   // per-wave P tile [q][key]

  const int tid  = threadIdx.x;
  const int lane = tid & 31;
  const int wave = tid >> 5;
  const int hi   = (lane >> 4) & 1;
  const int l15  = lane & 15;

  const int bx = blockIdx.x;
  const int qb = bx & 15;          // T/128 = 16 q-blocks
  const int h  = (bx >> 4) & 7;
  const int b  = bx >> 7;

  const int q0 = qb * 128 + wave * 16;
  const size_t ldq = 1536;
  const bf16_t* base = qkv + (size_t)(b * T_) * ldq + h * 64;

  // Q fragments for this wave's 16 rows (D = 64 -> two k-steps)
  const bf16_t* qp = base + (size_t)(q0 + l15) * ldq + hi * 16;
  const v16bf qa0 = load_a_frag(qp);
  const v16bf qa1 = load_a_frag(qp + 32);

  v8f o0 = (v8f){0,0,0,0,0,0,0,0}, o1 = o0, o2 = o0, o3 = o0;
  float m[8], s[8];
#pragma unroll
  for (int e = 0; e < 8; ++e) { m[e] = -1e30f; s[e] = 0.0f; }

  const float scale = 0.022097086912079608f;  // 2048^-0.5
  const int nj = qb * 4 + 4;                  // 32-key blocks up to diagonal

  const int r_ld  = tid >> 3;         // 0..31 key rows (cooperative load)
  const int c8_ld = (tid & 7) * 8;    // 8-elem chunks over D
  const unsigned ks_lds = (unsigned)(uintptr_t)&Ks[r_ld * 64 + c8_ld];

  for (int j = 0; j < nj; ++j) {
    const int s0 = j * 32;
    {
      // K tile: async DMA global -> LDS (b128 per lane), no VGPR round-trip
      const bf16_t* kp = base + (size_t)(s0 + r_ld) * ldq + 512 + c8_ld;
      asm volatile("global_load_async_to_lds_b128 %0, %1, off"
                   :: "v"(ks_lds), "v"(kp)
                   : "memory");
      // V tile: load through VGPRs and transpose into Vt while DMA runs
      union { uint4 q; bf16_t e[8]; } vv;
      vv.q = *(const uint4*)(base + (size_t)(s0 + r_ld) * ldq + 1024 + c8_ld);
#pragma unroll
      for (int i = 0; i < 8; ++i) Vt[(c8_ld + i) * 32 + r_ld] = vv.e[i];
      asm volatile("s_wait_asynccnt 0" ::: "memory");
    }
    __syncthreads();

    if (s0 <= q0 + 15) {  // skip fully-masked blocks for this wave
      // S = Q * K^T  (two 16x16 tiles covering 32 keys)
      v8f sa = (v8f){0,0,0,0,0,0,0,0}, sb = sa;
      const bf16_t* kb0 = &Ks[l15 * 64 + hi * 16];
      const bf16_t* kb1 = &Ks[(16 + l15) * 64 + hi * 16];
      sa = wmma_bf16(qa0, load_b_frag(kb0), sa);
      sa = wmma_bf16(qa1, load_b_frag(kb0 + 32), sa);
      sb = wmma_bf16(qa0, load_b_frag(kb1), sb);
      sb = wmma_bf16(qa1, load_b_frag(kb1 + 32), sb);

      const int key0 = s0 + l15;
      const int key1 = s0 + 16 + l15;
      bf16_t* pw = &Ps[wave][0];
#pragma unroll
      for (int e = 0; e < 8; ++e) {
        const int t = q0 + e + hi * 8;
        float v0 = sa[e] * scale; if (key0 > t) v0 = -1e30f;
        float v1 = sb[e] * scale; if (key1 > t) v1 = -1e30f;
        float mx = fmaxf(v0, v1);
        mx = fmaxf(mx, __shfl_xor(mx, 1, 32));
        mx = fmaxf(mx, __shfl_xor(mx, 2, 32));
        mx = fmaxf(mx, __shfl_xor(mx, 4, 32));
        mx = fmaxf(mx, __shfl_xor(mx, 8, 32));
        const float nm = fmaxf(m[e], mx);
        const float p0 = __expf(v0 - nm);
        const float p1 = __expf(v1 - nm);
        float rs = p0 + p1;
        rs += __shfl_xor(rs, 1, 32);
        rs += __shfl_xor(rs, 2, 32);
        rs += __shfl_xor(rs, 4, 32);
        rs += __shfl_xor(rs, 8, 32);
        const float alpha = __expf(m[e] - nm);
        m[e] = nm;
        s[e] = s[e] * alpha + rs;
        o0[e] *= alpha; o1[e] *= alpha; o2[e] *= alpha; o3[e] *= alpha;
        const int pr = e + hi * 8;
        pw[pr * 32 + l15]      = (bf16_t)p0;
        pw[pr * 32 + 16 + l15] = (bf16_t)p1;
      }
      // O += P * V   (P is A-operand, K-dim = 32 keys)
      const v16bf pa = load_a_frag(&pw[l15 * 32 + hi * 16]);
      o0 = wmma_bf16(pa, load_b_frag(&Vt[( 0 + l15) * 32 + hi * 16]), o0);
      o1 = wmma_bf16(pa, load_b_frag(&Vt[(16 + l15) * 32 + hi * 16]), o1);
      o2 = wmma_bf16(pa, load_b_frag(&Vt[(32 + l15) * 32 + hi * 16]), o2);
      o3 = wmma_bf16(pa, load_b_frag(&Vt[(48 + l15) * 32 + hi * 16]), o3);
    }
    __syncthreads();
  }

  bf16_t* ob = obuf + (size_t)(b * T_) * 512 + h * 64;
#pragma unroll
  for (int e = 0; e < 8; ++e) {
    const float inv = 1.0f / s[e];
    const int t = q0 + e + hi * 8;
    bf16_t* orow = ob + (size_t)t * 512 + l15;
    orow[0]  = (bf16_t)(o0[e] * inv);
    orow[16] = (bf16_t)(o1[e] * inv);
    orow[32] = (bf16_t)(o2[e] * inv);
    orow[48] = (bf16_t)(o3[e] * inv);
  }
}

// ---------------------------------------------------------------------------
extern "C" void kernel_launch(void* const* d_in, const int* in_sizes, int n_in,
                              void* d_out, int out_size, void* d_ws, size_t ws_size,
                              hipStream_t stream) {
  const float* x   = (const float*)d_in[0];
  const float* Wq  = (const float*)d_in[1];
  const float* Wk  = (const float*)d_in[2];
  const float* Wv  = (const float*)d_in[3];
  const float* Wo  = (const float*)d_in[4];
  const float* bo  = (const float*)d_in[5];
  const float* W1  = (const float*)d_in[6];
  const float* b1  = (const float*)d_in[7];
  const float* W2  = (const float*)d_in[8];
  const float* b2  = (const float*)d_in[9];
  const float* g1  = (const float*)d_in[10];
  const float* be1 = (const float*)d_in[11];
  const float* g2  = (const float*)d_in[12];
  const float* be2 = (const float*)d_in[13];
  float* out = (float*)d_out;

  // Workspace layout (~161 MB total)
  char* ws = (char*)d_ws;
  size_t off = 0;
  auto alloc = [&](size_t bytes) -> void* {
    void* p = ws + off;
    off += (bytes + 255) & ~(size_t)255;
    return p;
  };
  bf16_t* WQKVT = (bf16_t*)alloc((size_t)1536 * 512 * 2);
  bf16_t* WOT   = (bf16_t*)alloc((size_t)512 * 512 * 2);
  bf16_t* W1T   = (bf16_t*)alloc((size_t)4096 * 512 * 2);
  bf16_t* W2T   = (bf16_t*)alloc((size_t)512 * 4096 * 2);
  float*  XNF   = (float*) alloc((size_t)M_ * E_ * 4);
  bf16_t* XNB   = (bf16_t*)alloc((size_t)M_ * E_ * 2);
  bf16_t* QKV   = (bf16_t*)alloc((size_t)M_ * 1536 * 2);
  bf16_t* OBUF  = (bf16_t*)alloc((size_t)M_ * 512 * 2);
  float*  XN2F  = (float*) alloc((size_t)M_ * E_ * 4);
  bf16_t* XN2B  = (bf16_t*)alloc((size_t)M_ * E_ * 2);
  bf16_t* HBUF  = (bf16_t*)alloc((size_t)M_ * HID_ * 2);
  (void)ws_size; (void)in_sizes; (void)n_in; (void)out_size;

  // 0) weight conversion (bf16, N-major transposed)
  cvt_qkv_w<<<(3 * 512 * 512) / 256, 256, 0, stream>>>(Wq, Wk, Wv, WQKVT);
  transpose_cvt<<<(512 * 512) / 256, 256, 0, stream>>>(Wo, WOT, 512, 512);
  transpose_cvt<<<(512 * 4096) / 256, 256, 0, stream>>>(W1, W1T, 4096, 512);
  transpose_cvt<<<(512 * 4096) / 256, 256, 0, stream>>>(W2, W2T, 512, 4096);

  // 1) xn = LN(x)
  ln_kernel<<<M_ / 8, 256, 0, stream>>>(x, g1, be1, XNF, XNB);

  // 2) qkv = xn @ Wqkv                         (M=8192, N=1536, K=512)
  gemm_kernel<0, 0, 0, 1><<<dim3(1536 / 64, M_ / 128), 128, 0, stream>>>(
      XNB, WQKVT, nullptr, nullptr, nullptr, QKV, 512, 512, 512, 1536);

  // 3) o = causal_attention(q, k, v)
  attn_kernel<<<B_ * H_ * (T_ / 128), 256, 0, stream>>>(QKV, OBUF);

  // 4) x2 = xn + o @ Wo + bo   (in-place into XNF)
  gemm_kernel<1, 0, 1, 0><<<dim3(512 / 64, M_ / 128), 128, 0, stream>>>(
      OBUF, WOT, bo, XNF, XNF, nullptr, 512, 512, 512, 512);

  // 5) xn2 = LN(x2)
  ln_kernel<<<M_ / 8, 256, 0, stream>>>(XNF, g2, be2, XN2F, XN2B);

  // 6) h = relu(xn2 @ W1 + b1)                 (M=8192, N=4096, K=512)
  gemm_kernel<1, 1, 0, 1><<<dim3(4096 / 64, M_ / 128), 128, 0, stream>>>(
      XN2B, W1T, b1, nullptr, nullptr, HBUF, 512, 512, 512, 4096);

  // 7) out = xn2 + h @ W2 + b2                 (M=8192, N=512, K=4096)
  gemm_kernel<1, 0, 1, 0><<<dim3(512 / 64, M_ / 128), 128, 0, stream>>>(
      HBUF, W2T, b2, XN2F, out, nullptr, 4096, 4096, 4096, 512);
}
